// WindowAttention_59373627899919
// MI455X (gfx1250) — compile-verified, matrix operand order checked by reading, not run on hardware
//
#include <hip/hip_runtime.h>

typedef _Float16 h16;
typedef __attribute__((ext_vector_type(16))) _Float16 v16h;
typedef __attribute__((ext_vector_type(8)))  _Float16 v8h;
typedef __attribute__((ext_vector_type(8)))  float    v8f;

union V16U { v16h v; v8h h[2]; };

#define C_     192
#define C3_    576
#define HEADS_ 6
#define HD_    32
#define NTOK   64
#define HH_    256

// LDS byte offsets (all 16B aligned).
// region0: y f16[64][200] (phase1-2)  /  o f32[64][200] (phase3-4)  union
#define OFF_Y      0
#define OFF_O      0
#define OFF_QKV    51200                  // qkv f16 [64][584]  = 74752 B
#define OFF_QN     (OFF_QKV + 74752)      // qn  f16 [64][40]   = 5120 B
#define OFF_KN     (OFF_QN + 5120)        // kn  f16 [64][40]   = 5120 B
#define OFF_VT     (OFF_KN + 5120)        // vT  f16 [32][72]   = 4608 B
#define OFF_SC     (OFF_VT + 4608)        // sc  f32 [64][68]   = 17408 B
#define OFF_PR     (OFF_SC + 17408)       // pr  f16 [64][72]   = 9216 B
#define SMEM_TOTAL (OFF_PR + 9216)        // 167424 bytes

__global__ void convert_pw_f16(const float* __restrict__ w, h16* __restrict__ o, int n) {
    int i = blockIdx.x * 256 + threadIdx.x;
    if (i < n) o[i] = (h16)w[i];
}

// A fragment (16x32 f16, M rows): lane l<16 -> row r0+l, K chunks [k0..k0+7],[k0+16..k0+23]
// lanes 16..31 -> same rows, K chunks [k0+8..],[k0+24..].  B fragment mirrors with N in
// place of M and data stored K-contiguous per column.
__device__ __forceinline__ v16h load_frag(const h16* base, int rowStride,
                                          int r0, int k0, int l, int hi) {
    V16U f;
    const h16* p = base + (r0 + l) * rowStride + k0 + hi * 8;
    f.h[0] = *(const v8h*)(p);
    f.h[1] = *(const v8h*)(p + 16);
    return f.v;
}

__global__ __launch_bounds__(256) void win_attn_fused(
    const float* __restrict__ x, const float* __restrict__ dw_w,
    const float* __restrict__ dw_b, const h16* __restrict__ wf16,
    const float* __restrict__ pw_b, float* __restrict__ out)
{
    extern __shared__ char smem[];
    h16*   yL  = (h16*)  (smem + OFF_Y);
    float* oL  = (float*)(smem + OFF_O);
    h16*   qkv = (h16*)  (smem + OFF_QKV);
    h16*   qn  = (h16*)  (smem + OFF_QN);
    h16*   kn  = (h16*)  (smem + OFF_KN);
    h16*   vT  = (h16*)  (smem + OFF_VT);
    float* sc  = (float*)(smem + OFF_SC);
    h16*   pr  = (h16*)  (smem + OFF_PR);

    const int tid  = threadIdx.x;
    const int lane = tid & 31;
    const int wave = tid >> 5;
    const int l    = lane & 15;
    const int hi   = lane >> 4;

    const int blk = blockIdx.x;
    const int b   = blk >> 10;
    const int wy  = (blk >> 5) & 31;
    const int wx  = blk & 31;
    const int h0  = wy << 3;
    const int w0  = wx << 3;

    // ---- Phase 1: depthwise 3x3 (pad 1) + bias -> y f16 [token][channel] ----
    for (int i = tid; i < NTOK * C_; i += 256) {
        int c = i >> 6, t = i & 63;
        int ty = h0 + (t >> 3), tx = w0 + (t & 7);
        const float* xb  = x + ((b * C_ + c) << 16);
        const float* wdw = dw_w + c * 9;
        __builtin_prefetch(xb + (ty << 8) + tx, 0, 1);   // global_prefetch_b8
        float acc = dw_b[c];
        #pragma unroll
        for (int dy = -1; dy <= 1; ++dy) {
            int yy = ty + dy;
            if (yy < 0 || yy >= HH_) continue;
            #pragma unroll
            for (int dx = -1; dx <= 1; ++dx) {
                int xx = tx + dx;
                if (xx < 0 || xx >= HH_) continue;
                acc += xb[(yy << 8) + xx] * wdw[(dy + 1) * 3 + (dx + 1)];
            }
        }
        yL[t * 200 + c] = (h16)acc;
    }
    __syncthreads();

    // ---- Phase 2: pointwise GEMM qkv[64][576] = y[64][192] x W^T (+bias) ----
    for (int tIdx = wave; tIdx < 144; tIdx += 8) {
        int mt = tIdx & 3, nt = tIdx >> 2;
        v8f acc = {};
        #pragma unroll
        for (int kt = 0; kt < 6; ++kt) {
            v16h a  = load_frag(yL,   200, mt * 16, kt * 32, l, hi);
            v16h bm = load_frag(wf16, C_,  nt * 16, kt * 32, l, hi);
            acc = __builtin_amdgcn_wmma_f32_16x16x32_f16(false, a, false, bm,
                                                         (short)0, acc, false, false);
        }
        float bias = pw_b[nt * 16 + l];
        #pragma unroll
        for (int r = 0; r < 8; ++r) {
            int row = mt * 16 + r + hi * 8;
            qkv[row * 584 + nt * 16 + l] = (h16)(acc[r] + bias);
        }
    }
    __syncthreads();

    // ---- Phase 3: per-head cosine attention ----
    for (int h = 0; h < HEADS_; ++h) {
        // 3a: extract + L2-normalize q,k; transpose v
        if (tid < 128) {
            int t = tid & 63;
            const h16* src = qkv + t * 584 + (tid < 64 ? h * HD_ : C_ + h * HD_);
            h16* dst = (tid < 64 ? qn : kn) + t * 40;
            float ss = 0.f;
            #pragma unroll
            for (int d = 0; d < HD_; ++d) { float v = (float)src[d]; ss += v * v; }
            float rn = 1.f / fmaxf(sqrtf(ss), 1e-12f);
            #pragma unroll
            for (int d = 0; d < HD_; ++d) dst[d] = (h16)((float)src[d] * rn);
        }
        for (int i = tid; i < HD_ * NTOK; i += 256) {
            int d = i >> 6, t = i & 63;
            vT[d * 72 + t] = qkv[t * 584 + 2 * C_ + h * HD_ + d];
        }
        __syncthreads();

        // 3b: scores = qn . kn^T  (K = hd = 32 -> one WMMA per 16x16 tile)
        for (int tIdx = wave; tIdx < 16; tIdx += 8) {
            int mt = tIdx >> 2, nt = tIdx & 3;
            v16h a  = load_frag(qn, 40, mt * 16, 0, l, hi);
            v16h bm = load_frag(kn, 40, nt * 16, 0, l, hi);
            v8f acc = {};
            acc = __builtin_amdgcn_wmma_f32_16x16x32_f16(false, a, false, bm,
                                                         (short)0, acc, false, false);
            #pragma unroll
            for (int r = 0; r < 8; ++r)
                sc[(mt * 16 + r + hi * 8) * 68 + nt * 16 + l] = acc[r];
        }
        __syncthreads();

        // 3c: row softmax (f32) -> probs f16
        if (tid < 64) {
            const float* row = sc + tid * 68;
            float m = row[0];
            for (int j = 1; j < 64; ++j) m = fmaxf(m, row[j]);
            float s = 0.f;
            for (int j = 0; j < 64; ++j) s += __expf(row[j] - m);
            float inv = 1.f / s;
            h16* pd = pr + tid * 72;
            for (int j = 0; j < 64; ++j) pd[j] = (h16)(__expf(row[j] - m) * inv);
        }
        __syncthreads();

        // 3d: o = probs x v   (M=64, N=32, K=64 -> 8 tiles, one per wave)
        {
            int mt = wave >> 1, nt = wave & 1;
            v8f acc = {};
            #pragma unroll
            for (int kt = 0; kt < 2; ++kt) {
                v16h a  = load_frag(pr, 72, mt * 16, kt * 32, l, hi);
                v16h bm = load_frag(vT, 72, nt * 16, kt * 32, l, hi);
                acc = __builtin_amdgcn_wmma_f32_16x16x32_f16(false, a, false, bm,
                                                             (short)0, acc, false, false);
            }
            #pragma unroll
            for (int r = 0; r < 8; ++r)
                oL[(mt * 16 + r + hi * 8) * 200 + h * HD_ + nt * 16 + l] = acc[r];
        }
        __syncthreads();
    }

    // ---- Phase 4: window reverse -> BCHW f32 ----
    for (int i = tid; i < NTOK * C_; i += 256) {
        int c = i >> 6, t = i & 63;
        out[((b * C_ + c) << 16) + ((h0 + (t >> 3)) << 8) + (w0 + (t & 7))] =
            oL[t * 200 + c];
    }
}

extern "C" void kernel_launch(void* const* d_in, const int* in_sizes, int n_in,
                              void* d_out, int out_size, void* d_ws, size_t ws_size,
                              hipStream_t stream) {
    const float* x    = (const float*)d_in[0];
    const float* dw_w = (const float*)d_in[1];
    const float* dw_b = (const float*)d_in[2];
    const float* pw_w = (const float*)d_in[3];
    const float* pw_b = (const float*)d_in[4];
    float* out = (float*)d_out;
    h16* wf16 = (h16*)d_ws;                       // 576*192 f16 = 221 KB scratch

    int nw = C3_ * C_;
    convert_pw_f16<<<(nw + 255) / 256, 256, 0, stream>>>(pw_w, wf16, nw);
    win_attn_fused<<<4096, 256, SMEM_TOTAL, stream>>>(x, dw_w, dw_b, wf16, pw_b, out);
}
